// HybridGNN_51891794870980
// MI455X (gfx1250) — matrix operand order repaired and checked by reading, not compiled, not run on hardware
//
#include <hip/hip_runtime.h>
#include <hip/hip_bf16.h>

// ---------------------------------------------------------------------------
// HybridGNN forward for MI455X (gfx1250).
//  - Dense GEMMs: bf16 WMMA (v_wmma_f32_16x16x32_bf16), f32 accumulate,
//    K templated (full unroll), 4 N-tiles per wave (A-fragment reuse),
//    weights pre-packed into WMMA register order -> all b128 loads.
//  - Graph aggregation: wave-per-edge coalesced gather + f32 global atomics
//  - GAT softmax: two-pass (atomicMax on order-preserving uint encoding).
// ---------------------------------------------------------------------------

#define CDIV(a,b) (((a)+(b)-1)/(b))
typedef unsigned int u32;

typedef __attribute__((ext_vector_type(16))) __bf16 v16bf;
typedef __attribute__((ext_vector_type(8)))  float  v8f;

__device__ __forceinline__ float lrelu02(float v){ return v > 0.f ? v : 0.2f * v; }
__device__ __forceinline__ unsigned fenc(float f){
  unsigned u = __float_as_uint(f);
  return (u & 0x80000000u) ? ~u : (u | 0x80000000u);
}
__device__ __forceinline__ float fdec(unsigned u){
  return __uint_as_float((u & 0x80000000u) ? (u & 0x7fffffffu) : ~u);
}
__device__ __forceinline__ unsigned short bf16bits(float f){
  __hip_bfloat16 h = __float2bfloat16(f);
  unsigned short s; __builtin_memcpy(&s, &h, 2); return s;
}

// ---------------------------------------------------------------------------
// Pack fp32 weight W[K,Nc] into WMMA-B register order (bf16 pairs):
//   out[((blk*Nc + col)*16 + 8*half + r)] = bf16(W[kb,col]) | bf16(W[kb+1,col])<<16
//   with kb = blk*32 + 16*half + 2*r   (contiguous k = 16*half + s; A uses the same)
// ---------------------------------------------------------------------------
__global__ void k_packB(const float* __restrict__ W, u32* __restrict__ out, int K, int Nc){
  long t = blockIdx.x * (long)blockDim.x + threadIdx.x;
  long total = (long)(K / 32) * Nc * 16;
  if (t >= total) return;
  int idx = (int)(t & 15); long q = t >> 4;
  int col = (int)(q % Nc); int blk = (int)(q / Nc);
  int half = idx >> 3, r = idx & 7;
  int kb = blk * 32 + 16 * half + 2 * r;
  unsigned lo = bf16bits(W[(long)kb * Nc + col]);
  unsigned hi = bf16bits(W[(long)(kb + 1) * Nc + col]);
  out[t] = lo | (hi << 16);
}

// ---------------------------------------------------------------------------
// GEMM: C[M,Nc] = A[M,K](bf16 row-major) * Bp(packed) (+bias) (+=C if accum)
// One wave -> 16 rows x NT*16 cols. A fragment per lane = two contiguous 16B
// chunks (global_load_b128 x2); B fragment = two b128 from the packed layout.
// ---------------------------------------------------------------------------
template<int K, int NT>
__global__ void k_gemm_wmma(const __hip_bfloat16* __restrict__ A,
                            const u32* __restrict__ Bp,
                            const float* __restrict__ bias,
                            float* __restrict__ C,
                            int M, int Nc, int accum)
{
  int wave = (int)((blockIdx.x * (long)blockDim.x + threadIdx.x) >> 5);
  int lane = threadIdx.x & 31;
  int groups = Nc / (16 * NT);
  int tiles  = (M >> 4) * groups;
  if (wave >= tiles) return;                 // wave-uniform: EXEC stays all-1s
  int tm = wave / groups, tg = wave % groups;
  int m = lane & 15, half = lane >> 4;
  int rowA = tm * 16 + m;

  v8f c[NT];
  if (accum) {
#pragma unroll
    for (int t = 0; t < NT; ++t)
#pragma unroll
      for (int r = 0; r < 8; ++r)
        c[t][r] = C[(long)(tm*16 + r + 8*half) * Nc + (tg*NT + t)*16 + m];
  } else {
#pragma unroll
    for (int t = 0; t < NT; ++t)
#pragma unroll
      for (int r = 0; r < 8; ++r) c[t][r] = 0.f;
  }
  float bv[NT];
#pragma unroll
  for (int t = 0; t < NT; ++t) bv[t] = bias ? bias[(tg*NT + t)*16 + m] : 0.f;

  const u32* Arow = reinterpret_cast<const u32*>(A + (long)rowA * K);
#pragma unroll
  for (int blk = 0; blk < K / 32; ++blk) {
    v16bf a;
    uint4* av = reinterpret_cast<uint4*>(&a);
    av[0] = *reinterpret_cast<const uint4*>(Arow + blk*16 + 8*half);      // K=k0+16g..+7
    av[1] = *reinterpret_cast<const uint4*>(Arow + blk*16 + 8*half + 4);  // K=k0+16g+8..
#pragma unroll
    for (int t = 0; t < NT; ++t) {
      int colB = (tg*NT + t)*16 + m;
      const u32* bp = Bp + ((long)blk * Nc + colB) * 16 + 8*half;
      v16bf b;
      uint4* bvv = reinterpret_cast<uint4*>(&b);
      bvv[0] = *reinterpret_cast<const uint4*>(bp);
      bvv[1] = *reinterpret_cast<const uint4*>(bp + 4);
      c[t] = __builtin_amdgcn_wmma_f32_16x16x32_bf16(false, a, false, b,
                                                     (short)0, c[t], false, false);
    }
  }

#pragma unroll
  for (int t = 0; t < NT; ++t)
#pragma unroll
    for (int r = 0; r < 8; ++r)
      C[(long)(tm*16 + r + 8*half) * Nc + (tg*NT + t)*16 + m] = c[t][r] + bv[t];
}

// ---------------------------------------------------------------------------
// Utility kernels
// ---------------------------------------------------------------------------
__global__ void k_f2bf(const float* __restrict__ in, __hip_bfloat16* __restrict__ out, long n){
  long i = blockIdx.x * (long)blockDim.x + threadIdx.x;
  if (i < n) out[i] = __float2bfloat16(in[i]);
}
__global__ void k_fillf(float* __restrict__ p, float v, long n){
  long i = blockIdx.x * (long)blockDim.x + threadIdx.x;
  if (i < n) p[i] = v;
}
__global__ void k_count(const int* __restrict__ d, float* __restrict__ deg, int E){
  int e = blockIdx.x * blockDim.x + threadIdx.x;
  if (e < E) atomicAdd(&deg[d[e]], 1.f);
}
__global__ void k_rsqrt_deg(const float* __restrict__ deg, float* __restrict__ dinv, int Nn){
  int i = blockIdx.x * blockDim.x + threadIdx.x;
  if (i < Nn) { float v = deg[i]; dinv[i] = v > 0.f ? rsqrtf(v) : 0.f; }
}

// BN (+ optional activation) -> bf16 slice (row stride ldbf, col offset) and/or f32
__global__ void k_bn_act(const float* __restrict__ in, int M, int F,
                         const float* __restrict__ g, const float* __restrict__ be,
                         const float* __restrict__ mn, const float* __restrict__ vr,
                         int act, __hip_bfloat16* __restrict__ obf, int ldbf, int offbf,
                         float* __restrict__ of, int ldf)
{
  long t = blockIdx.x * (long)blockDim.x + threadIdx.x;
  if (t >= (long)M * F) return;
  int n = (int)(t / F), f = (int)(t % F);
  float v = in[t];
  v = (v - mn[f]) * rsqrtf(vr[f] + 1e-5f) * g[f] + be[f];
  if (act == 1)      v = fmaxf(v, 0.f);
  else if (act == 2) v = v > 0.f ? v : (expf(v) - 1.f);
  if (obf) obf[(long)n * ldbf + offbf + f] = __float2bfloat16(v);
  if (of)  of[(long)n * ldf + f] = v;
}

// ---------------------------------------------------------------------------
// GCN:  out = D^-1/2 (A+I) D^-1/2 (xW) + b
// ---------------------------------------------------------------------------
__global__ void k_gcn_init(const float* __restrict__ xw, const float* __restrict__ dinv,
                           const float* __restrict__ bias, float* __restrict__ out, int Nn){
  long t = blockIdx.x * (long)blockDim.x + threadIdx.x;
  if (t >= (long)Nn * 128) return;
  int n = (int)(t >> 7), f = (int)(t & 127);
  float di = dinv[n];
  out[t] = xw[t] * di * di + bias[f];       // self-loop term + bias
}
__global__ void k_gcn_scatter(const int* __restrict__ src, const int* __restrict__ dst,
                              const float* __restrict__ xw, const float* __restrict__ dinv,
                              float* __restrict__ out, int E){
  long gid = blockIdx.x * (long)blockDim.x + threadIdx.x;
  int wave = (int)(gid >> 5), lane = (int)(gid & 31);
  if (wave >= E) return;
  int s = src[wave], d = dst[wave];
  float cf = dinv[s] * dinv[d];
  const float* xr = xw + (long)s * 128;
  float* orow = out + (long)d * 128;
#pragma unroll
  for (int j = 0; j < 4; ++j) { int f = lane + 32 * j; atomicAdd(&orow[f], xr[f] * cf); }
}

// ---------------------------------------------------------------------------
// GAT
// ---------------------------------------------------------------------------
__global__ void k_gat_alpha(const float* __restrict__ xw, const float* __restrict__ attS,
                            const float* __restrict__ attD,
                            float* __restrict__ as_, float* __restrict__ ad_,
                            int Nn, int H, int C){
  int t = blockIdx.x * blockDim.x + threadIdx.x;
  if (t >= Nn * H) return;
  int n = t / H, h = t % H;
  const float* row = xw + (long)n * H * C + (long)h * C;
  float s = 0.f, d = 0.f;
  for (int c = 0; c < C; ++c) { float v = row[c]; s += v * attS[h*C + c]; d += v * attD[h*C + c]; }
  as_[t] = s; ad_[t] = d;
}
__global__ void k_gat_minit(const float* __restrict__ as_, const float* __restrict__ ad_,
                            unsigned* __restrict__ menc, int NH){
  int t = blockIdx.x * blockDim.x + threadIdx.x;
  if (t < NH) menc[t] = fenc(lrelu02(as_[t] + ad_[t]));  // self-loop logit seeds the max
}
__global__ void k_gat_edgemax(const int* __restrict__ src, const int* __restrict__ dst,
                              const float* __restrict__ as_, const float* __restrict__ ad_,
                              unsigned* __restrict__ menc, int E, int H){
  int t = blockIdx.x * blockDim.x + threadIdx.x;
  if (t >= E * H) return;
  int e = t / H, h = t % H;
  int s = src[e], d = dst[e];
  float logit = lrelu02(as_[s*H + h] + ad_[d*H + h]);
  atomicMax(&menc[d*H + h], fenc(logit));
}
__global__ void k_gat_zinit(const float* __restrict__ xw, const float* __restrict__ as_,
                            const float* __restrict__ ad_, const unsigned* __restrict__ menc,
                            float* __restrict__ z, float* __restrict__ out,
                            int Nn, int H, int C){
  long t = blockIdx.x * (long)blockDim.x + threadIdx.x;
  if (t >= (long)Nn * H * C) return;
  int n = (int)(t / (H * C)), f = (int)(t % (H * C)), h = f / C, c = f % C;
  float sl = lrelu02(as_[n*H + h] + ad_[n*H + h]);
  float e  = expf(sl - fdec(menc[n*H + h]));
  out[t] = xw[t] * e;                       // self-loop numerator
  if (c == 0) z[n*H + h] = e;               // self-loop denominator seed
}
__global__ void k_gat_edgescatter(const int* __restrict__ src, const int* __restrict__ dst,
                                  const float* __restrict__ xw, const float* __restrict__ as_,
                                  const float* __restrict__ ad_, const unsigned* __restrict__ menc,
                                  float* __restrict__ z, float* __restrict__ out,
                                  int E, int H, int C){
  long gid = blockIdx.x * (long)blockDim.x + threadIdx.x;
  int wave = (int)(gid >> 5), lane = (int)(gid & 31);
  if (wave >= E * H) return;
  int e = wave / H, h = wave % H;
  int s = src[e], d = dst[e];
  float logit = lrelu02(as_[s*H + h] + ad_[d*H + h]);
  float ee = expf(logit - fdec(menc[d*H + h]));
  if (lane == 0) atomicAdd(&z[d*H + h], ee);
  const float* xr = xw + ((long)s * H + h) * C;
  float* orow = out + ((long)d * H + h) * C;
  for (int c = lane; c < C; c += 32) atomicAdd(&orow[c], xr[c] * ee);
}
__global__ void k_gat_fin(const float* __restrict__ num, const float* __restrict__ z,
                          const float* __restrict__ bias, int M, int H, int C,
                          const float* __restrict__ g, const float* __restrict__ be,
                          const float* __restrict__ mn, const float* __restrict__ vr,
                          int act, __hip_bfloat16* __restrict__ obf, int ldbf, int offbf){
  long t = blockIdx.x * (long)blockDim.x + threadIdx.x;
  if (t >= (long)M * H * C) return;
  int n = (int)(t / (H * C)), f = (int)(t % (H * C)), h = f / C;
  float v = num[t] / (z[n*H + h] + 1e-16f) + bias[f];
  v = (v - mn[f]) * rsqrtf(vr[f] + 1e-5f) * g[f] + be[f];
  if (act == 2) v = v > 0.f ? v : (expf(v) - 1.f);
  obf[(long)n * ldbf + offbf + f] = __float2bfloat16(v);
}

// ---------------------------------------------------------------------------
// SAGE
// ---------------------------------------------------------------------------
__global__ void k_sage_scatter(const int* __restrict__ src, const int* __restrict__ dst,
                               const float* __restrict__ h, float* __restrict__ agg, int E){
  long gid = blockIdx.x * (long)blockDim.x + threadIdx.x;
  int wave = (int)(gid >> 5), lane = (int)(gid & 31);
  if (wave >= E) return;
  int s = src[wave], d = dst[wave];
  const float* xr = h + (long)s * 128;
  float* orow = agg + (long)d * 128;
#pragma unroll
  for (int j = 0; j < 4; ++j) { int f = lane + 32 * j; atomicAdd(&orow[f], xr[f]); }
}
__global__ void k_sage_mean(const float* __restrict__ agg, const float* __restrict__ cnt,
                            __hip_bfloat16* __restrict__ aggbf, int Nn){
  long t = blockIdx.x * (long)blockDim.x + threadIdx.x;
  if (t >= (long)Nn * 128) return;
  int n = (int)(t >> 7);
  aggbf[t] = __float2bfloat16(agg[t] / fmaxf(cnt[n], 1.f));
}

// ---------------------------------------------------------------------------
// Classifier: [N,128] @ [128,2] + b
// ---------------------------------------------------------------------------
__global__ void k_cls(const float* __restrict__ fused, const float* __restrict__ W,
                      const float* __restrict__ b, float* __restrict__ out, int Nn){
  int n = blockIdx.x * blockDim.x + threadIdx.x;
  if (n >= Nn) return;
  const float* fr = fused + (long)n * 128;
  float a0 = b[0], a1 = b[1];
  for (int k = 0; k < 128; ++k) { float v = fr[k]; a0 += v * W[2*k]; a1 += v * W[2*k + 1]; }
  out[2*n] = a0; out[2*n + 1] = a1;
}

// ---------------------------------------------------------------------------
extern "C" void kernel_launch(void* const* d_in, const int* in_sizes, int n_in,
                              void* d_out, int out_size, void* d_ws, size_t ws_size,
                              hipStream_t stream)
{
  (void)n_in; (void)out_size; (void)ws_size;
  const int D = 128, BT = 256;
  const int N = in_sizes[0] / D;
  const int E = in_sizes[1] / 2;
  const float* x  = (const float*)d_in[0];
  const int* src  = (const int*)d_in[1];
  const int* dst  = src + E;

  // ---- parameter pointers (setup_inputs() dict insertion order, DFS) ----
  const float *gcnW[3], *gcnB[3], *gcnBN[3][4];
  for (int i = 0; i < 3; ++i) { int b0 = 2 + 6*i;
    gcnW[i] = (const float*)d_in[b0]; gcnB[i] = (const float*)d_in[b0+1];
    for (int j = 0; j < 4; ++j) gcnBN[i][j] = (const float*)d_in[b0+2+j]; }
  const float *gatW[3], *gatAS[3], *gatAD[3], *gatB[3], *gatBN[3][4];
  for (int i = 0; i < 3; ++i) { int b0 = 20 + 8*i;
    gatW[i] = (const float*)d_in[b0]; gatAS[i] = (const float*)d_in[b0+1];
    gatAD[i] = (const float*)d_in[b0+2]; gatB[i] = (const float*)d_in[b0+3];
    for (int j = 0; j < 4; ++j) gatBN[i][j] = (const float*)d_in[b0+4+j]; }
  const float *sgWl[3], *sgBl[3], *sgWr[3], *sgBN[3][4];
  for (int i = 0; i < 3; ++i) { int b0 = 44 + 7*i;
    sgWl[i] = (const float*)d_in[b0]; sgBl[i] = (const float*)d_in[b0+1];
    sgWr[i] = (const float*)d_in[b0+2];
    for (int j = 0; j < 4; ++j) sgBN[i][j] = (const float*)d_in[b0+3+j]; }
  const float* fuW = (const float*)d_in[65]; const float* fuB = (const float*)d_in[66];
  const float* fBN[4] = {(const float*)d_in[67], (const float*)d_in[68],
                         (const float*)d_in[69], (const float*)d_in[70]};
  const float* clW = (const float*)d_in[71]; const float* clB = (const float*)d_in[72];

  // ---- workspace bump allocator ----
  char* wsp = (char*)d_ws; size_t off = 0;
  auto alloc = [&](size_t bytes) -> void* {
    void* p = wsp + off; off = (off + bytes + 255) & ~(size_t)255; return p; };
  __hip_bfloat16* xbf    = (__hip_bfloat16*)alloc((size_t)N*D*2);
  __hip_bfloat16* abf    = (__hip_bfloat16*)alloc((size_t)N*D*2);
  __hip_bfloat16* bbf    = (__hip_bfloat16*)alloc((size_t)N*D*2);
  __hip_bfloat16* aggbf  = (__hip_bfloat16*)alloc((size_t)N*D*2);
  __hip_bfloat16* combbf = (__hip_bfloat16*)alloc((size_t)N*320*2);
  u32 *wg[3], *wa[3], *wl[3], *wr[3];                       // packed weights
  for (int i = 0; i < 3; ++i) wg[i] = (u32*)alloc(128*128*2);
  for (int i = 0; i < 3; ++i) wa[i] = (u32*)alloc(128*((i<2)?128:64)*2);
  for (int i = 0; i < 3; ++i) { wl[i] = (u32*)alloc(128*128*2);
                                wr[i] = (u32*)alloc(128*128*2); }
  u32* wfu = (u32*)alloc(320*128*2);
  float* xw    = (float*)alloc((size_t)N*D*4);
  float* outb  = (float*)alloc((size_t)N*D*4);
  float* aggf  = (float*)alloc((size_t)N*D*4);
  float* sact0 = (float*)alloc((size_t)N*D*4);
  float* sact1 = (float*)alloc((size_t)N*D*4);
  float* deg   = (float*)alloc((size_t)N*4);
  float* dinv  = (float*)alloc((size_t)N*4);
  float* cnt   = (float*)alloc((size_t)N*4);
  float* asr   = (float*)alloc((size_t)N*2*4);
  float* ads   = (float*)alloc((size_t)N*2*4);
  unsigned* me = (unsigned*)alloc((size_t)N*2*4);
  float* zs    = (float*)alloc((size_t)N*2*4);
  float* fout  = (float*)d_out;           // [N,2] logits then [N,128] fused
  float* fused = fout + (long)2*N;

  auto conv = [&](const float* in, __hip_bfloat16* o, long n){
    k_f2bf<<<CDIV(n, BT), BT, 0, stream>>>(in, o, n); };
  auto pack = [&](const float* W, u32* o, int K, int Nc){
    long tot = (long)K * Nc / 2;
    k_packB<<<CDIV(tot, BT), BT, 0, stream>>>(W, o, K, Nc); };
  auto gemm = [&](const __hip_bfloat16* A, const u32* Bp, const float* bias,
                  float* C, int M, int K, int Nc, int acc){
    int tiles = (M >> 4) * (Nc / 64);                      // NT=4 -> 64 cols/wave
    if (K == 128)
      k_gemm_wmma<128,4><<<CDIV(tiles, BT/32), BT, 0, stream>>>(A, Bp, bias, C, M, Nc, acc);
    else
      k_gemm_wmma<320,4><<<CDIV(tiles, BT/32), BT, 0, stream>>>(A, Bp, bias, C, M, Nc, acc);
  };
  auto bnact = [&](const float* in, int F, const float* const* bn, int act,
                   __hip_bfloat16* obf, int ldbf, int offbf, float* of, int ldf){
    k_bn_act<<<CDIV((long)N*F, BT), BT, 0, stream>>>(in, N, F, bn[0], bn[1], bn[2], bn[3],
                                                     act, obf, ldbf, offbf, of, ldf); };

  // ---- one-time conversions + degrees ----
  conv(x, xbf, (long)N*D);
  for (int i = 0; i < 3; ++i) pack(gcnW[i], wg[i], 128, 128);
  for (int i = 0; i < 3; ++i) pack(gatW[i], wa[i], 128, (i<2)?128:64);
  for (int i = 0; i < 3; ++i) { pack(sgWl[i], wl[i], 128, 128); pack(sgWr[i], wr[i], 128, 128); }
  pack(fuW, wfu, 320, 128);
  k_fillf<<<CDIV(N, BT), BT, 0, stream>>>(deg, 1.f, N);     // self loop
  k_count<<<CDIV(E, BT), BT, 0, stream>>>(dst, deg, E);
  k_rsqrt_deg<<<CDIV(N, BT), BT, 0, stream>>>(deg, dinv, N);
  k_fillf<<<CDIV(N, BT), BT, 0, stream>>>(cnt, 0.f, N);     // SAGE: no self loop
  k_count<<<CDIV(E, BT), BT, 0, stream>>>(dst, cnt, E);

  // ---- GCN branch ----
  { const __hip_bfloat16* h = xbf;
    for (int i = 0; i < 3; ++i) {
      gemm(h, wg[i], nullptr, xw, N, 128, 128, 0);
      k_gcn_init<<<CDIV((long)N*128, BT), BT, 0, stream>>>(xw, dinv, gcnB[i], outb, N);
      k_gcn_scatter<<<CDIV((long)E*32, BT), BT, 0, stream>>>(src, dst, xw, dinv, outb, E);
      if (i < 2) { __hip_bfloat16* o = (i == 0) ? abf : bbf;
        bnact(outb, 128, gcnBN[i], 1, o, 128, 0, nullptr, 0); h = o; }
      else bnact(outb, 128, gcnBN[i], 0, combbf, 320, 0, nullptr, 0);
    } }

  // ---- GAT branch ----
  { const __hip_bfloat16* h = xbf;
    for (int i = 0; i < 3; ++i) {
      int H = (i < 2) ? 2 : 1, C = 64, HC = H * C;
      gemm(h, wa[i], nullptr, xw, N, 128, HC, 0);
      k_gat_alpha<<<CDIV(N*H, BT), BT, 0, stream>>>(xw, gatAS[i], gatAD[i], asr, ads, N, H, C);
      k_gat_minit<<<CDIV(N*H, BT), BT, 0, stream>>>(asr, ads, me, N*H);
      k_gat_edgemax<<<CDIV(E*H, BT), BT, 0, stream>>>(src, dst, asr, ads, me, E, H);
      k_gat_zinit<<<CDIV((long)N*HC, BT), BT, 0, stream>>>(xw, asr, ads, me, zs, outb, N, H, C);
      k_gat_edgescatter<<<CDIV((long)E*H*32, BT), BT, 0, stream>>>(src, dst, xw, asr, ads,
                                                                   me, zs, outb, E, H, C);
      if (i < 2) { __hip_bfloat16* o = (i == 0) ? abf : bbf;
        k_gat_fin<<<CDIV((long)N*HC, BT), BT, 0, stream>>>(outb, zs, gatB[i], N, H, C,
            gatBN[i][0], gatBN[i][1], gatBN[i][2], gatBN[i][3], 2, o, 128, 0); h = o; }
      else
        k_gat_fin<<<CDIV((long)N*HC, BT), BT, 0, stream>>>(outb, zs, gatB[i], N, H, C,
            gatBN[i][0], gatBN[i][1], gatBN[i][2], gatBN[i][3], 0, combbf, 320, 128);
    } }

  // ---- SAGE branch ----
  { const float* hf = x; const __hip_bfloat16* h = xbf;
    for (int i = 0; i < 3; ++i) {
      k_fillf<<<CDIV((long)N*128, BT), BT, 0, stream>>>(aggf, 0.f, (long)N*128);
      k_sage_scatter<<<CDIV((long)E*32, BT), BT, 0, stream>>>(src, dst, hf, aggf, E);
      k_sage_mean<<<CDIV((long)N*128, BT), BT, 0, stream>>>(aggf, cnt, aggbf, N);
      gemm(aggbf, wl[i], sgBl[i], outb, N, 128, 128, 0);
      gemm(h, wr[i], nullptr, outb, N, 128, 128, 1);        // += x @ Wr
      if (i < 2) { __hip_bfloat16* o = (i == 0) ? abf : bbf;
        float* sf = (i == 0) ? sact0 : sact1;
        bnact(outb, 128, sgBN[i], 1, o, 128, 0, sf, 128); h = o; hf = sf; }
      else bnact(outb, 128, sgBN[i], 0, combbf, 320, 192, nullptr, 0);
    } }

  // ---- fusion + classifier ----
  gemm(combbf, wfu, fuB, xw, N, 320, 128, 0);
  bnact(xw, 128, fBN, 1, nullptr, 0, 0, fused, 128);
  k_cls<<<CDIV(N, BT), BT, 0, stream>>>(fused, clW, clB, fout, N);
}
